// CRF_25855703121938
// MI455X (gfx1250) — matrix operand order repaired and checked by reading, not compiled, not run on hardware
//
#include <hip/hip_runtime.h>
#include <hip/hip_bf16.h>
#include <math.h>

typedef __attribute__((ext_vector_type(16))) _Float16 v16h;
typedef __attribute__((ext_vector_type(8)))  _Float16 v8h;
typedef __attribute__((ext_vector_type(4)))  _Float16 v4h;
typedef __attribute__((ext_vector_type(2)))  __fp16   v2hf;   // return type of cvt_pkrtz
typedef __attribute__((ext_vector_type(8)))  float    v8f;
typedef __attribute__((ext_vector_type(4)))  float    v4f;

#define Bdim 128
#define Tdim 512
#define Ndim 256
#define EPITCH 264   // halfs; 528B row stride -> 4-bank skew per row (conflict-free B frags)
#define PPITCH 264
#define SPITCH 256   // floats
#define PFDIST 6     // prefetch distance (steps) for the emissions stream

// ---------------------------------------------------------------------------
// Forward algorithm. One workgroup per 16 batch rows (M=16 WMMA tile).
// Per step: score'[m][j] = emit[t,m,j] + M[m] + log( P[m,:] (f16) @ E[:,j] (f16) )
// with P = exp(score - rowmax), E = exp(trans) staged transposed in LDS.
// ---------------------------------------------------------------------------
__global__ __launch_bounds__(256)
void crf_forward_kernel(const float* __restrict__ emissions,
                        const float* __restrict__ trans,
                        float* __restrict__ z)
{
    extern __shared__ char smem_raw[];
    _Float16* E   = (_Float16*)smem_raw;            // [Ndim][EPITCH] : E_t[n][k]=exp(trans[k][n])
    _Float16* P   = E + Ndim * EPITCH;              // [16][PPITCH]
    float*    S   = (float*)(P + 16 * PPITCH);      // [16][SPITCH]
    float*    Mrw = S + 16 * SPITCH;                // [16]

    const int tid  = threadIdx.x;
    const int wave = tid >> 5;
    const int lane = tid & 31;
    const int b0   = blockIdx.x * 16;

    // Stage E = exp(trans), transposed for contiguous-K B-fragment loads.
    for (int idx = tid; idx < Ndim * Ndim; idx += 256) {
        int i = idx >> 8;          // from-state (K)
        int j = idx & 255;         // to-state   (N)
        E[j * EPITCH + i] = (_Float16)__expf(trans[idx]);
    }
    // Initial score = emissions[:, 0, :]
    for (int idx = tid; idx < 16 * Ndim; idx += 256) {
        int m = idx >> 8;
        int j = idx & 255;
        S[m * SPITCH + j] = emissions[(size_t)(b0 + m) * Tdim * Ndim + j];
    }
    __syncthreads();

    const int m16 = tid >> 4;      // row owned in max/exp phase
    const int l16 = tid & 15;
    const int c4  = l16 * 4;       // interleaved group base: cols c4 + 64*i, i=0..3
    const int h   = lane >> 4;     // wave half (WMMA layout)
    const int nl  = lane & 15;
    const int n0  = wave * 32;     // this wave's 32-column slice

    // Software-pipelined emissions tile: emitc = step t, emitn = step t+1.
    float emitc[16], emitn[16];
    {
        const float* ebase = emissions + ((size_t)b0 * Tdim + 1) * Ndim;
        #pragma unroll
        for (int r = 0; r < 8; ++r) {
            size_t row = (size_t)(r + 8 * h) * Tdim * Ndim;
            emitc[r]     = ebase[row + n0 + nl];
            emitc[r + 8] = ebase[row + n0 + 16 + nl];
        }
    }

    for (int t = 1; t < Tdim; ++t) {
        // Issue next step's emissions loads (consumed next iteration).
        // Clamped instead of guarded -> branch-free body.
        {
            int tn = (t + 1 < Tdim) ? (t + 1) : (Tdim - 1);
            const float* ebase = emissions + ((size_t)b0 * Tdim + tn) * Ndim;
            #pragma unroll
            for (int r = 0; r < 8; ++r) {
                size_t row = (size_t)(r + 8 * h) * Tdim * Ndim;
                emitn[r]     = ebase[row + n0 + nl];
                emitn[r + 8] = ebase[row + n0 + 16 + nl];
            }
        }
        // Prefetch the t+PFDIST tile into cache (global_prefetch_b8).
        {
            int tp = (t + PFDIST < Tdim) ? (t + PFDIST) : (Tdim - 1);
            if (tid < 128) {
                const float* pfb = emissions + ((size_t)b0 * Tdim + tp) * Ndim
                                 + (size_t)(tid >> 3) * Tdim * Ndim + (tid & 7) * 32;
                __builtin_prefetch(pfb, 0, 1);
            }
        }

        // (a)+(b): per-thread interleaved groups (cols c4+64i) -> rowmax + packed exp.
        // At load i, threads hit 16 distinct bank-quads -> conflict-free, and all
        // register indices are compile-time constants.
        float v[16];
        #pragma unroll
        for (int i = 0; i < 4; ++i) {
            v4f x = *(const v4f*)&S[m16 * SPITCH + c4 + 64 * i];
            v[4 * i + 0] = x[0]; v[4 * i + 1] = x[1];
            v[4 * i + 2] = x[2]; v[4 * i + 3] = x[3];
        }
        float mx = v[0];
        #pragma unroll
        for (int i = 1; i < 16; ++i) mx = fmaxf(mx, v[i]);
        #pragma unroll
        for (int o = 8; o >= 1; o >>= 1) mx = fmaxf(mx, __shfl_xor(mx, o, 16));
        if (l16 == 0) Mrw[m16] = mx;
        #pragma unroll
        for (int i = 0; i < 4; ++i) {
            v2hf a = __builtin_amdgcn_cvt_pkrtz(__expf(v[4 * i + 0] - mx),
                                                __expf(v[4 * i + 1] - mx));
            v2hf b = __builtin_amdgcn_cvt_pkrtz(__expf(v[4 * i + 2] - mx),
                                                __expf(v[4 * i + 3] - mx));
            v4h pk;
            pk[0] = (_Float16)a[0]; pk[1] = (_Float16)a[1];
            pk[2] = (_Float16)b[0]; pk[3] = (_Float16)b[1];
            *(v4h*)&P[m16 * PPITCH + c4 + 64 * i] = pk;
        }
        __syncthreads();

        // (c) GEMM slice: C(16x32) = P(16x256) @ E(256x256)[:, n0:n0+32]
        v8f c0 = {}; v8f c1 = {};
        #pragma unroll
        for (int kb = 0; kb < 8; ++kb) {
            const int kbase = kb * 32;
            // A fragment (16-bit A 16x32 layout): lane m=nl, half h
            v16h a;
            {
                const v8h lo = *(const v8h*)&P[nl * PPITCH + kbase + 16 * h];
                const v8h hi = *(const v8h*)&P[nl * PPITCH + kbase + 16 * h + 8];
                #pragma unroll
                for (int c = 0; c < 8; ++c) { a[c] = lo[c]; a[c + 8] = hi[c]; }
            }
            // B fragments: lane n=nl(+tile), K = kbase + 16h + c (contiguous in E_t row)
            v16h bf0, bf1;
            {
                const _Float16* eb0 = &E[(n0 + nl) * EPITCH + kbase + 16 * h];
                const v8h x0 = *(const v8h*)(eb0);
                const v8h x1 = *(const v8h*)(eb0 + 8);
                const _Float16* eb1 = &E[(n0 + 16 + nl) * EPITCH + kbase + 16 * h];
                const v8h y0 = *(const v8h*)(eb1);
                const v8h y1 = *(const v8h*)(eb1 + 8);
                #pragma unroll
                for (int c = 0; c < 8; ++c) {
                    bf0[c] = x0[c]; bf0[c + 8] = x1[c];
                    bf1[c] = y0[c]; bf1[c + 8] = y1[c];
                }
            }
            c0 = __builtin_amdgcn_wmma_f32_16x16x32_f16(false, a, false, bf0,
                                                        (short)0, c0, false, false);
            c1 = __builtin_amdgcn_wmma_f32_16x16x32_f16(false, a, false, bf1,
                                                        (short)0, c1, false, false);
        }

        // (d) score update (D layout: VGPR r -> row r+8h, col = lane&15)
        {
            v4f mr0 = *(const v4f*)&Mrw[8 * h];
            v4f mr1 = *(const v4f*)&Mrw[8 * h + 4];
            #pragma unroll
            for (int r = 0; r < 8; ++r) {
                int m = r + 8 * h;
                float base = (r < 4) ? mr0[r & 3] : mr1[r & 3];
                S[m * SPITCH + n0 + nl]      = emitc[r]     + base + __logf(c0[r]);
                S[m * SPITCH + n0 + 16 + nl] = emitc[r + 8] + base + __logf(c1[r]);
            }
        }
        #pragma unroll
        for (int i = 0; i < 16; ++i) emitc[i] = emitn[i];
        __syncthreads();
    }

    // Final z[b] = logsumexp over states (same interleaved mapping)
    {
        float v[16];
        #pragma unroll
        for (int i = 0; i < 4; ++i) {
            v4f x = *(const v4f*)&S[m16 * SPITCH + c4 + 64 * i];
            v[4 * i + 0] = x[0]; v[4 * i + 1] = x[1];
            v[4 * i + 2] = x[2]; v[4 * i + 3] = x[3];
        }
        float mx = v[0];
        #pragma unroll
        for (int i = 1; i < 16; ++i) mx = fmaxf(mx, v[i]);
        #pragma unroll
        for (int o = 8; o >= 1; o >>= 1) mx = fmaxf(mx, __shfl_xor(mx, o, 16));
        float s = 0.f;
        #pragma unroll
        for (int i = 0; i < 16; ++i) s += __expf(v[i] - mx);
        #pragma unroll
        for (int o = 8; o >= 1; o >>= 1) s += __shfl_xor(s, o, 16);
        if (l16 == 0) z[b0 + m16] = mx + __logf(s);
    }
}

// ---------------------------------------------------------------------------
// Gold path score: s0 + sum_t (emit_gather + trans_gather) * mask
// ---------------------------------------------------------------------------
__global__ __launch_bounds__(256)
void crf_gold_kernel(const float* __restrict__ emissions,
                     const int*   __restrict__ tags,
                     const float* __restrict__ mask,
                     const float* __restrict__ trans,
                     float* __restrict__ g)
{
    __shared__ float red[256];
    const int b   = blockIdx.x;
    const int tid = threadIdx.x;
    const int*   tg = tags + (size_t)b * Tdim;
    const float* em = emissions + (size_t)b * Tdim * Ndim;
    const float* mk = mask + (size_t)b * Tdim;

    float acc = 0.f;
    for (int t = 1 + tid; t < Tdim; t += 256) {
        int cur = tg[t], prev = tg[t - 1];
        acc += (em[(size_t)t * Ndim + cur] + trans[prev * Ndim + cur]) * mk[t];
    }
    red[tid] = acc;
    __syncthreads();
    for (int o = 128; o >= 1; o >>= 1) {
        if (tid < o) red[tid] += red[tid + o];
        __syncthreads();
    }
    if (tid == 0) g[b] = red[0] + em[tg[0]];
}

// ---------------------------------------------------------------------------
// out = mean(z - g)
// ---------------------------------------------------------------------------
__global__ void crf_final_kernel(const float* __restrict__ z,
                                 const float* __restrict__ g,
                                 float* __restrict__ out)
{
    __shared__ float red[128];
    int tid = threadIdx.x;
    red[tid] = z[tid] - g[tid];
    __syncthreads();
    for (int o = 64; o >= 1; o >>= 1) {
        if (tid < o) red[tid] += red[tid + o];
        __syncthreads();
    }
    if (tid == 0) out[0] = red[0] * (1.0f / (float)Bdim);
}

extern "C" void kernel_launch(void* const* d_in, const int* in_sizes, int n_in,
                              void* d_out, int out_size, void* d_ws, size_t ws_size,
                              hipStream_t stream) {
    const float* emissions = (const float*)d_in[0];
    const int*   tags      = (const int*)d_in[1];   // harness passes integer inputs as int32
    const float* mask      = (const float*)d_in[2];
    const float* trans     = (const float*)d_in[3];

    float* z = (float*)d_ws;        // [Bdim]
    float* g = z + Bdim;            // [Bdim]

    size_t smem = (size_t)(Ndim * EPITCH + 16 * PPITCH) * sizeof(_Float16)
                + (size_t)(16 * SPITCH + 16) * sizeof(float);   // ~160 KB, fits 320 KB LDS/WGP

    crf_forward_kernel<<<Bdim / 16, 256, smem, stream>>>(emissions, trans, z);
    crf_gold_kernel<<<Bdim, 256, 0, stream>>>(emissions, tags, mask, trans, g);
    crf_final_kernel<<<1, 128, 0, stream>>>(z, g, (float*)d_out);
}
